// MultiHeadAttention_1821066134205
// MI455X (gfx1250) — matrix operand order repaired and checked by reading, not compiled, not run on hardware
//
#include <hip/hip_runtime.h>

// MHA forward for MI455X (gfx1250, wave32, WMMA).
// Pipeline: f32->bf16 convert -> 3 projection GEMMs (bf16 WMMA, f32 acc)
//           -> flash attention (bf16 WMMA) -> output GEMM (f32 out).
//
// Workspace layout (needs 64 MB):
//   [0,8M)    query bf16          [8M,16M)  key bf16      [16M,24M) value bf16
//   [24M,26M) Wq bf16  [26M,28M) Wk bf16  [28M,30M) Wv bf16  [30M,32M) Wo bf16
//   [32M,40M) Q  [B,H,S,Dk] bf16
//   [40M,48M) K  [B,H,S,Dk] bf16
//   [48M,56M) Vt [B,H,Dk,S] bf16   (V pre-transposed for P*V WMMA B-fragments)
//   [56M,64M) attn [B,S,1024] bf16

typedef unsigned short u16;
typedef __attribute__((ext_vector_type(16))) __bf16 v16bf;
typedef __attribute__((ext_vector_type(8)))  float  v8f;
typedef int i32x4 __attribute__((vector_size(16)));   // matches builtin param type

#define D_MODEL 1024
#define N_HEADS 16
#define D_K     64
#define SEQ     2048
#define BATCH   2
#define MROWS   (BATCH * SEQ)   // 4096

#define AS1 __attribute__((address_space(1)))
#define AS3 __attribute__((address_space(3)))

#if defined(__has_builtin)
#if __has_builtin(__builtin_amdgcn_global_load_async_to_lds_b128) && \
    __has_builtin(__builtin_amdgcn_s_wait_asynccnt)
#define HAVE_ASYNC_LDS 1
#endif
#endif
#ifndef HAVE_ASYNC_LDS
#define HAVE_ASYNC_LDS 0
#endif

// 16B global -> LDS tile copy. Async path uses the CDNA5 DMA-to-LDS
// instruction (ASYNCcnt-tracked); fallback is a VGPR round trip.
__device__ __forceinline__ void ld_tile128(u16* lds_dst, const u16* gsrc) {
#if HAVE_ASYNC_LDS
    __builtin_amdgcn_global_load_async_to_lds_b128(
        (AS1 i32x4*)(unsigned long long)gsrc,
        (AS3 i32x4*)(unsigned)(unsigned long long)lds_dst,  // ISA: LDS = addr[31:0]
        0, 0);
#else
    *(uint4*)lds_dst = *(const uint4*)gsrc;
#endif
}

__device__ __forceinline__ void async_join() {
#if HAVE_ASYNC_LDS
    __builtin_amdgcn_s_wait_asynccnt(0);
#endif
}

__device__ __forceinline__ u16 f2bf(float f) {
    unsigned u = __float_as_uint(f);
    unsigned r = u + 0x7FFFu + ((u >> 16) & 1u);   // round to nearest even
    return (u16)(r >> 16);
}

union FragBF { uint4 q[2]; v16bf v; };

// A-matrix 16x32 bf16 fragment (ISA 7.12.2): m = lane&15, half = lane>>4.
// Lanes 0-15: VGPR0-3 hold k=0..7, VGPR4-7 hold k=16..23.
// Lanes 16-31: k=8..15 and k=24..31.  => two contiguous 16B runs per lane.
__device__ __forceinline__ v16bf load_frag_a(const u16* base, int ld) {
    const int lane = threadIdx.x & 31;
    const int m = lane & 15, h = lane >> 4;
    FragBF f;
    f.q[0] = *(const uint4*)(base + m * ld + 8 * h);
    f.q[1] = *(const uint4*)(base + m * ld + 16 + 8 * h);
    return f.v;
}

// B-matrix 32x16 bf16 fragment, read from B^T storage (row n, contiguous k):
// lane n = lane&15; lanes 0-15 hold k=0..7,16..23, lanes 16-31 hold k=8..15,24..31.
__device__ __forceinline__ v16bf load_frag_b(const u16* base, int ld) {
    const int lane = threadIdx.x & 31;
    const int n = lane & 15, h = lane >> 4;
    FragBF f;
    f.q[0] = *(const uint4*)(base + n * ld + 8 * h);
    f.q[1] = *(const uint4*)(base + n * ld + 16 + 8 * h);
    return f.v;
}

__device__ __forceinline__ v8f wmma_bf16(v16bf a, v16bf b, v8f c) {
    return __builtin_amdgcn_wmma_f32_16x16x32_bf16(
        false, a, false, b, (short)0, c, false, false);
}

// ---------------------------------------------------------------- convert
__global__ __launch_bounds__(256) void f32_to_bf16_kernel(
    const float* __restrict__ src, u16* __restrict__ dst, int n4) {
    int i = blockIdx.x * 256 + threadIdx.x;
    if (i < n4) {
        float4 f = ((const float4*)src)[i];
        uint2 p;
        p.x = (unsigned)f2bf(f.x) | ((unsigned)f2bf(f.y) << 16);
        p.y = (unsigned)f2bf(f.z) | ((unsigned)f2bf(f.w) << 16);
        ((uint2*)dst)[i] = p;
    }
}

// ---------------------------------------------------------------- GEMM
// Y[m][n] = sum_k X[m][k] * W[n][k] + bias[n]
// Block tile: 64(M) x 256(N) x 32(K), 256 threads = 8 waves (2x4),
// each wave computes a 32x64 sub-tile = 2x4 WMMA tiles (8 WMMA / k-step).
// MODE 0/1: write bf16 -> [B,H,S,Dk] ; MODE 2: bf16 -> [B,H,Dk,S] ;
// MODE 3: write f32 row-major (+bias) to d_out.  (compile-time MODE ->
// branch-free epilogues)
template <int MODE>
__global__ __launch_bounds__(256) void gemm_qkvo(
    const u16* __restrict__ X, const u16* __restrict__ W,
    const float* __restrict__ bias, void* __restrict__ out) {
    __shared__ u16 As[64][40];     // 40-halfword stride: 80B rows, 16B aligned
    __shared__ u16 Bs[256][40];

    const int t = threadIdx.x;
    const int lane = t & 31, wave = t >> 5;
    const int wm = wave >> 2, wn = wave & 3;           // 2 x 4 wave grid
    const int blockM = blockIdx.y * 64, blockN = blockIdx.x * 256;
    const int n = lane & 15, half = lane >> 4;

    const int arow = t >> 2, achunk = (t & 3) * 8;     // 64 rows x 4 chunks
    const u16* Xrow = X + (blockM + arow) * D_MODEL + achunk;
    const u16* Wrow = W + (blockN + t) * D_MODEL;      // one W row per thread

    v8f c[2][4] = {};

    for (int k0 = 0; k0 < D_MODEL; k0 += 32) {
        __syncthreads();
        ld_tile128(&As[arow][achunk], Xrow + k0);
        #pragma unroll
        for (int jc = 0; jc < 4; ++jc)
            ld_tile128(&Bs[t][jc * 8], Wrow + k0 + jc * 8);
        async_join();
        __syncthreads();

        v16bf a[2], b[4];
        a[0] = load_frag_a(&As[wm * 32][0],      40);
        a[1] = load_frag_a(&As[wm * 32 + 16][0], 40);
        #pragma unroll
        for (int j = 0; j < 4; ++j)
            b[j] = load_frag_b(&Bs[wn * 64 + j * 16][0], 40);
        #pragma unroll
        for (int i = 0; i < 2; ++i)
            #pragma unroll
            for (int j = 0; j < 4; ++j)
                c[i][j] = wmma_bf16(a[i], b[j], c[i][j]);
    }

    // Epilogue. C layout: row m = v + 8*half, col n = lane&15.
    #pragma unroll
    for (int i = 0; i < 2; ++i)
        #pragma unroll
        for (int j = 0; j < 4; ++j)
            #pragma unroll
            for (int v = 0; v < 8; ++v) {
                const int gm = blockM + wm * 32 + i * 16 + v + 8 * half;
                const int gn = blockN + wn * 64 + j * 16 + n;
                const float val = c[i][j][v] + bias[gn];
                if (MODE == 3) {
                    ((float*)out)[gm * D_MODEL + gn] = val;
                } else {
                    const int b = gm >> 11, s = gm & (SEQ - 1);
                    const int h = gn >> 6,  d = gn & (D_K - 1);
                    const int idx = (MODE == 2)
                        ? (((b * N_HEADS + h) * D_K) + d) * SEQ + s    // Vt: [B,H,Dk,S]
                        : (((b * N_HEADS + h) * SEQ) + s) * D_K + d;   // Q/K: [B,H,S,Dk]
                    ((u16*)out)[idx] = f2bf(val);
                }
            }
}

// ---------------------------------------------------------------- attention
// One block = one (b,h) and 128 query rows; 8 waves x 16 rows.
// Flash (online-softmax) loop over kv in steps of 32.
__global__ __launch_bounds__(256) void mha_attn(
    const u16* __restrict__ Q, const u16* __restrict__ K,
    const u16* __restrict__ Vt, u16* __restrict__ attn) {
    __shared__ u16 Ks[32][72];        // [kv][d], 144B rows
    __shared__ u16 Vts[64][40];       // [d][kv], 80B rows
    __shared__ u16 Ps[8][16][40];     // per-wave P transpose bounce

    const int t = threadIdx.x, lane = t & 31, wave = t >> 5;
    const int bh = blockIdx.y;                 // 0..31
    const int b = bh >> 4, h = bh & 15;
    const int qbase = blockIdx.x * 128 + wave * 16;
    const int n = lane & 15, half = lane >> 4;

    // Q fragments stay in registers: 16 rows x 64 (two k=32 steps).
    const u16* qptr = Q + (bh * SEQ + qbase) * D_K;
    const v16bf qa0 = load_frag_a(qptr, D_K);        // d = 0..31
    const v16bf qa1 = load_frag_a(qptr + 32, D_K);   // d = 32..63

    v8f o[4] = {};
    float mrow[8], lrow[8];
    #pragma unroll
    for (int v = 0; v < 8; ++v) { mrow[v] = -1e30f; lrow[v] = 0.f; }

    const int krow = t >> 3, kchunk = (t & 7) * 8;   // K tile: 32 x 64
    const int vrow = t >> 2, vchunk = (t & 3) * 8;   // Vt tile: 64 x 32

    for (int kv = 0; kv < SEQ; kv += 32) {
        __syncthreads();
        ld_tile128(&Ks[krow][kchunk],  K  + (bh * SEQ + kv + krow) * D_K + kchunk);
        ld_tile128(&Vts[vrow][vchunk], Vt + (bh * D_K + vrow) * SEQ + kv + vchunk);
        async_join();
        __syncthreads();

        // scores S = Q * K^T  (16 x 32), two 16-col tiles, k over d=64
        v8f s0 = {}, s1 = {};
        s0 = wmma_bf16(qa0, load_frag_b(&Ks[0][0],   72), s0);
        s0 = wmma_bf16(qa1, load_frag_b(&Ks[0][32],  72), s0);
        s1 = wmma_bf16(qa0, load_frag_b(&Ks[16][0],  72), s1);
        s1 = wmma_bf16(qa1, load_frag_b(&Ks[16][32], 72), s1);

        // online softmax, per row m = v + 8*half (stats reduced over 16 lanes)
        #pragma unroll
        for (int v = 0; v < 8; ++v) {
            float a0 = s0[v] * 0.125f;          // 1/sqrt(64)
            float a1 = s1[v] * 0.125f;
            float mv = fmaxf(a0, a1);
            #pragma unroll
            for (int off = 1; off < 16; off <<= 1)
                mv = fmaxf(mv, __shfl_xor(mv, off, 16));
            const float nm = fmaxf(mrow[v], mv);
            const float scale = __expf(mrow[v] - nm);
            const float p0 = __expf(a0 - nm);
            const float p1 = __expf(a1 - nm);
            float rs = p0 + p1;
            #pragma unroll
            for (int off = 1; off < 16; off <<= 1)
                rs += __shfl_xor(rs, off, 16);
            lrow[v] = lrow[v] * scale + rs;
            mrow[v] = nm;
            o[0][v] *= scale; o[1][v] *= scale;
            o[2][v] *= scale; o[3][v] *= scale;
            const int m = v + 8 * half;
            Ps[wave][m][n]      = f2bf(p0);
            Ps[wave][m][16 + n] = f2bf(p1);
        }

        // O += P * V   (B^T[n=d][k=kv] = Vts[d][kv], contiguous in kv)
        const v16bf pa = load_frag_a(&Ps[wave][0][0], 40);
        o[0] = wmma_bf16(pa, load_frag_b(&Vts[0][0],  40), o[0]);
        o[1] = wmma_bf16(pa, load_frag_b(&Vts[16][0], 40), o[1]);
        o[2] = wmma_bf16(pa, load_frag_b(&Vts[32][0], 40), o[2]);
        o[3] = wmma_bf16(pa, load_frag_b(&Vts[48][0], 40), o[3]);
    }

    // write attn[b][s][h*64 + d] as bf16 (input layout for output GEMM)
    #pragma unroll
    for (int dj = 0; dj < 4; ++dj)
        #pragma unroll
        for (int v = 0; v < 8; ++v) {
            const int m = v + 8 * half;
            const float val = o[dj][v] / lrow[v];
            const int s = qbase + m;
            const int col = h * D_K + dj * 16 + n;
            attn[(b * SEQ + s) * D_MODEL + col] = f2bf(val);
        }
}

// ---------------------------------------------------------------- launch
extern "C" void kernel_launch(void* const* d_in, const int* in_sizes, int n_in,
                              void* d_out, int out_size, void* d_ws, size_t ws_size,
                              hipStream_t stream) {
    const float* q  = (const float*)d_in[0];
    const float* k  = (const float*)d_in[1];
    const float* v  = (const float*)d_in[2];
    const float* Wq = (const float*)d_in[3];
    const float* bq = (const float*)d_in[4];
    const float* Wk = (const float*)d_in[5];
    const float* bk = (const float*)d_in[6];
    const float* Wv = (const float*)d_in[7];
    const float* bv = (const float*)d_in[8];
    const float* Wo = (const float*)d_in[9];
    const float* bo = (const float*)d_in[10];

    char* ws = (char*)d_ws;
    const size_t MB = 1024 * 1024;
    u16* qbf  = (u16*)(ws + 0 * MB);
    u16* kbf  = (u16*)(ws + 8 * MB);
    u16* vbf  = (u16*)(ws + 16 * MB);
    u16* wqbf = (u16*)(ws + 24 * MB);
    u16* wkbf = (u16*)(ws + 26 * MB);
    u16* wvbf = (u16*)(ws + 28 * MB);
    u16* wobf = (u16*)(ws + 30 * MB);
    u16* Qp   = (u16*)(ws + 32 * MB);
    u16* Kp   = (u16*)(ws + 40 * MB);
    u16* Vtp  = (u16*)(ws + 48 * MB);
    u16* atn  = (u16*)(ws + 56 * MB);

    const int n4in = MROWS * D_MODEL / 4;     // 1M vec4
    const int n4w  = D_MODEL * D_MODEL / 4;   // 256K vec4
    f32_to_bf16_kernel<<<n4in / 256, 256, 0, stream>>>(q,  qbf,  n4in);
    f32_to_bf16_kernel<<<n4in / 256, 256, 0, stream>>>(k,  kbf,  n4in);
    f32_to_bf16_kernel<<<n4in / 256, 256, 0, stream>>>(v,  vbf,  n4in);
    f32_to_bf16_kernel<<<n4w  / 256, 256, 0, stream>>>(Wq, wqbf, n4w);
    f32_to_bf16_kernel<<<n4w  / 256, 256, 0, stream>>>(Wk, wkbf, n4w);
    f32_to_bf16_kernel<<<n4w  / 256, 256, 0, stream>>>(Wv, wvbf, n4w);
    f32_to_bf16_kernel<<<n4w  / 256, 256, 0, stream>>>(Wo, wobf, n4w);

    dim3 ggrid(D_MODEL / 256, MROWS / 64);    // (4, 64)
    gemm_qkvo<0><<<ggrid, 256, 0, stream>>>(qbf, wqbf, bq, Qp);
    gemm_qkvo<1><<<ggrid, 256, 0, stream>>>(kbf, wkbf, bk, Kp);
    gemm_qkvo<2><<<ggrid, 256, 0, stream>>>(vbf, wvbf, bv, Vtp);

    mha_attn<<<dim3(SEQ / 128, BATCH * N_HEADS), 256, 0, stream>>>(Qp, Kp, Vtp, atn);

    gemm_qkvo<3><<<ggrid, 256, 0, stream>>>(atn, wobf, bo, d_out);
}
